// HGAT_1975684956198
// MI455X (gfx1250) — matrix operand order repaired and checked by reading, compiled, not run on hardware
//
#include <hip/hip_runtime.h>
#include <hip/hip_bf16.h>

typedef __attribute__((ext_vector_type(16))) _Float16 v16h;
typedef __attribute__((ext_vector_type(8)))  _Float16 v8h;
typedef __attribute__((ext_vector_type(8)))  float    v8f;

#define BB     4
#define NN     4096
#define CIN    128
#define COUT   64

// -------------------------------------------------------------------------
// Kernel 1: h = node_rep @ proj_W^T + proj_b   (WMMA f16 -> f32)
// Each wave computes a 16x64 tile of h. Also writes hT (f16, [b][o][n]).
// -------------------------------------------------------------------------
__global__ __launch_bounds__(256) void k_proj(const float* __restrict__ node_rep,
                                              const float* __restrict__ proj_W,
                                              const float* __restrict__ proj_b,
                                              float*    __restrict__ h,    // [B*N][64]
                                              _Float16* __restrict__ hT)   // [B][64][N]
{
    const int lane = threadIdx.x & 31;
    const int wave = threadIdx.x >> 5;
    const int tile = blockIdx.x * 8 + wave;      // 16-row tile over flattened B*N
    const int row0 = tile * 16;
    const int mrow = lane & 15;
    const int half = lane >> 4;

    v8f acc[4] = {};

    for (int kb = 0; kb < CIN; kb += 32) {
        // A fragment: 16x32, contiguous k order (same as B)
        // lanes 0-15: elems 0..15 -> K=kb+0..15
        // lanes16-31: elems 0..15 -> K=kb+16..31
        const float* arow = node_rep + (size_t)(row0 + mrow) * CIN + kb;
        v16h a;
        #pragma unroll
        for (int e = 0; e < 8; ++e) a[e]     = (_Float16)arow[half * 16 + e];
        #pragma unroll
        for (int e = 0; e < 8; ++e) a[8 + e] = (_Float16)arow[half * 16 + 8 + e];

        #pragma unroll
        for (int nt = 0; nt < 4; ++nt) {
            // B fragment: N = lane%16, K = kb + (lane/16)*16 + e ; B[k][n] = proj_W[n][k]
            const int   ncol = nt * 16 + mrow;
            const float* brow = proj_W + (size_t)ncol * CIN + kb + half * 16;
            v16h bf;
            #pragma unroll
            for (int e = 0; e < 16; ++e) bf[e] = (_Float16)brow[e];
            acc[nt] = __builtin_amdgcn_wmma_f32_16x16x32_f16(
                false, a, false, bf, (short)0, acc[nt], false, false);
        }
    }

    const int b  = row0 / NN;
    const int n0 = row0 % NN;
    #pragma unroll
    for (int nt = 0; nt < 4; ++nt) {
        const int   ncol = nt * 16 + mrow;
        const float bias = proj_b[ncol];
        #pragma unroll
        for (int r = 0; r < 8; ++r) {
            const int   m   = r + half * 8;       // C/D layout: M = r (+8 for hi lanes)
            const float val = acc[nt][r] + bias;
            h[(size_t)(row0 + m) * COUT + ncol] = val;
            hT[(size_t)b * COUT * NN + (size_t)ncol * NN + (n0 + m)] = (_Float16)val;
        }
    }
}

// -------------------------------------------------------------------------
// Kernel 2: per-node type-gathered scores  f1 = <h, k_W[t]> + k_b[t], f2 likewise
// -------------------------------------------------------------------------
__global__ __launch_bounds__(256) void k_scores(const float* __restrict__ h,
                                                const int*   __restrict__ ntype,
                                                const float* __restrict__ kW,
                                                const float* __restrict__ kb,
                                                const float* __restrict__ vW,
                                                const float* __restrict__ vb,
                                                float* __restrict__ f1,
                                                float* __restrict__ f2)
{
    const int idx = blockIdx.x * 256 + threadIdx.x;   // over B*N
    const int n   = idx % NN;
    const int t   = ntype[n];
    const float* hr = h  + (size_t)idx * COUT;
    const float* kr = kW + (size_t)t * COUT;
    const float* vr = vW + (size_t)t * COUT;
    float a1 = 0.f, a2 = 0.f;
    #pragma unroll 8
    for (int o = 0; o < COUT; ++o) { const float hv = hr[o]; a1 += hv * kr[o]; a2 += hv * vr[o]; }
    f1[idx] = a1 + kb[t];
    f2[idx] = a2 + vb[t];
}

// -------------------------------------------------------------------------
// Kernel 3: column softmax denominators (softmax over axis i for each (b,j))
// cinv[b,j] = 1 / sum_i exp(sigmoid(adj[i,j]*(f1[b,i]+f2[b,j])) - 0.5)
// -------------------------------------------------------------------------
__global__ __launch_bounds__(256) void k_colsum(const float* __restrict__ adj,
                                                const float* __restrict__ f1,
                                                const float* __restrict__ f2,
                                                float* __restrict__ cinv)
{
    const int idx = blockIdx.x * 256 + threadIdx.x;   // over B*N
    const int b   = idx / NN;
    const int j   = idx % NN;
    const float  f2c = f2[idx];
    const float* f1b = f1 + (size_t)b * NN;
    float acc = 0.f;
    #pragma unroll 4
    for (int i = 0; i < NN; ++i) {
        const float u = adj[(size_t)i * NN + j] * (f1b[i] + f2c);   // coalesced over j
        const float s = 1.f / (1.f + __expf(-u)) - 0.5f;
        acc += __expf(s);
    }
    cinv[idx] = 1.f / acc;
}

// -------------------------------------------------------------------------
// Kernel 4: out[b,i,:] = sum_j p[b,i,j] * h[b,j,:]  via WMMA f16->f32
// Wave computes a 16(i) x 64(o) tile; K = j in steps of 32 with the 16x32
// probability tile materialized in f16 LDS and regathered in A-fragment order.
// -------------------------------------------------------------------------
__global__ __launch_bounds__(256) void k_attn(const float*    __restrict__ adj,
                                              const float*    __restrict__ f1,
                                              const float*    __restrict__ f2,
                                              const float*    __restrict__ cinv,
                                              const _Float16* __restrict__ hT,  // [B][64][N]
                                              float*          __restrict__ out) // [B*N][64]
{
    __shared__ __align__(16) _Float16 pT[8][16 * 32];   // per-wave 16x32 f16 tile

    const int lane = threadIdx.x & 31;
    const int wave = threadIdx.x >> 5;
    const int tile = blockIdx.x * 8 + wave;
    const int row0 = tile * 16;
    const int b    = row0 / NN;
    const int n0   = row0 % NN;
    const int mrow = lane & 15;
    const int half = lane >> 4;

    const _Float16* hTb = hT + (size_t)b * COUT * NN;
    const float*    f2b = f2   + (size_t)b * NN;
    const float*    civ = cinv + (size_t)b * NN;

    float f1m[16];
    #pragma unroll
    for (int m = 0; m < 16; ++m) f1m[m] = f1[(size_t)b * NN + n0 + m];

    v8f acc[4] = {};
    _Float16* myT = pT[wave];

    for (int jb = 0; jb < NN; jb += 32) {
        // ---- build normalized probability tile: lane = column c in [0,32) ----
        const int   j   = jb + lane;
        const float f2c = f2b[j];
        const float ci  = civ[j];
        __builtin_prefetch(&adj[(size_t)(n0 + mrow) * NN + jb + 128], 0, 3); // global_prefetch_b8
        #pragma unroll 4
        for (int m = 0; m < 16; ++m) {
            const float u = adj[(size_t)(n0 + m) * NN + j] * (f1m[m] + f2c);
            const float s = 1.f / (1.f + __expf(-u)) - 0.5f;
            myT[m * 32 + lane] = (_Float16)(__expf(s) * ci);
        }
        __builtin_amdgcn_wave_barrier();
        asm volatile("s_wait_dscnt 0x0" ::: "memory");   // in-wave LDS write->read order

        // ---- A fragment from LDS (contiguous k = 16*half + s, same as B) ----
        const v8h lo = *(const v8h*)&myT[mrow * 32 + half * 16];
        const v8h hi = *(const v8h*)&myT[mrow * 32 + half * 16 + 8];
        const v16h a = __builtin_shufflevector(lo, hi, 0,1,2,3,4,5,6,7,8,9,10,11,12,13,14,15);

        // ---- B fragments: 16 contiguous f16 per lane from hT ----
        #pragma unroll
        for (int nt = 0; nt < 4; ++nt) {
            const v16h bf = *(const v16h*)&hTb[(size_t)(nt * 16 + mrow) * NN + jb + half * 16];
            acc[nt] = __builtin_amdgcn_wmma_f32_16x16x32_f16(
                false, a, false, bf, (short)0, acc[nt], false, false);
        }
        __builtin_amdgcn_wave_barrier();                 // WAR before next tile write
    }

    #pragma unroll
    for (int nt = 0; nt < 4; ++nt) {
        #pragma unroll
        for (int r = 0; r < 8; ++r) {
            const int m = r + half * 8;
            out[(size_t)(row0 + m) * COUT + nt * 16 + mrow] = acc[nt][r];
        }
    }
}

// -------------------------------------------------------------------------
extern "C" void kernel_launch(void* const* d_in, const int* in_sizes, int n_in,
                              void* d_out, int out_size, void* d_ws, size_t ws_size,
                              hipStream_t stream)
{
    const float* node_rep = (const float*)d_in[0];   // [B,N,128]
    const float* adj      = (const float*)d_in[1];   // [N,N]
    const int*   ntype    = (const int*)  d_in[2];   // [N]
    const float* proj_W   = (const float*)d_in[3];   // [64,128]
    const float* proj_b   = (const float*)d_in[4];   // [64]
    const float* k_W      = (const float*)d_in[5];   // [5,64]
    const float* k_b      = (const float*)d_in[6];   // [5]
    const float* v_W      = (const float*)d_in[7];   // [5,64]
    const float* v_b      = (const float*)d_in[8];   // [5]
    float*       out      = (float*)d_out;           // [B,N,64]

    // workspace carve-up (256B aligned blocks)
    char* w = (char*)d_ws;
    float*    h    = (float*)w;                                    // 4*4096*64*4 = 4 MiB
    _Float16* hT   = (_Float16*)(w + (size_t)BB * NN * COUT * 4);  // 2 MiB
    float*    f1   = (float*)(w + (size_t)BB * NN * COUT * 6);
    float*    f2   = f1 + (size_t)BB * NN;
    float*    cinv = f2 + (size_t)BB * NN;

    const int tiles = (BB * NN) / 16;            // 1024 sixteen-row tiles

    k_proj  <<<tiles / 8, 256, 0, stream>>>(node_rep, proj_W, proj_b, h, hT);
    k_scores<<<(BB * NN) / 256, 256, 0, stream>>>(h, ntype, k_W, k_b, v_W, v_b, f1, f2);
    k_colsum<<<(BB * NN) / 256, 256, 0, stream>>>(adj, f1, f2, cinv);
    k_attn  <<<tiles / 8, 256, 0, stream>>>(adj, f1, f2, cinv, hT, out);
}
